// Encoding_1786706395267
// MI455X (gfx1250) — matrix operand order repaired and checked by reading, not compiled, not run on hardware
//
#include <hip/hip_runtime.h>
#include <hip/hip_bf16.h>

typedef __attribute__((ext_vector_type(16))) _Float16 v16h;
typedef __attribute__((ext_vector_type(8)))  float    v8f;

#define ENC_B 8
#define ENC_D 64
#define ENC_K 32
#define ENC_N 3136          // 56*56
#define TILE_N 64
#define TILES_PER_B 49      // 3136 / 64

// ---------------------------------------------------------------------------
// Pass 1: per-(b,n,d) softmax over K, write E to d_out (B,D,H,W layout),
// and deterministic per-block pixel-sum partials for EM.
// Block = 256 threads: 4 d-groups x 64 pixels. Grid = 8*49.
// ---------------------------------------------------------------------------
__global__ __launch_bounds__(256) void enc_pass1(
    const float* __restrict__ X, const float* __restrict__ cw,
    const float* __restrict__ sc, float* __restrict__ E,
    float* __restrict__ partial)
{
    __shared__ float s_c[ENC_K * ENC_D];   // 8 KB codewords
    __shared__ float s_s[ENC_K * ENC_D];   // 8 KB scales
    __shared__ float s_red[ENC_D * 65];    // padded (65) to dodge bank conflicts

    const int t = threadIdx.x;
    for (int i = t; i < ENC_K * ENC_D; i += 256) {
        s_c[i] = cw[i];
        s_s[i] = sc[i];
    }
    __syncthreads();

    const int b    = blockIdx.x / TILES_PER_B;
    const int tile = blockIdx.x - b * TILES_PER_B;
    const int pix  = t & 63;      // pixel within tile (coalesced over lanes)
    const int dg   = t >> 6;      // d-group 0..3, 16 channels each
    const int n    = tile * TILE_N + pix;

    const float* Xb = X + (size_t)b * ENC_D * ENC_N;
    float*       Eb = E + (size_t)b * ENC_D * ENC_N;

    float tq[ENC_K];

    for (int dd = 0; dd < 16; ++dd) {
        const int d = dg * 16 + dd;
        const float x = Xb[d * ENC_N + n];

        // pass A: logits + running max (all lanes hit the same LDS addr -> broadcast)
        float m = -3.402823466e38f;
#pragma unroll
        for (int k = 0; k < ENC_K; ++k) {
            float r = x - s_c[k * ENC_D + d];
            float v = s_s[k * ENC_D + d] * r * r;
            tq[k] = v;
            m = fmaxf(m, v);
        }

        // pass B: exp + weighted residual sums
        float se = 0.f, ser = 0.f;
#pragma unroll
        for (int k = 0; k < ENC_K; ++k) {
            float a = __expf(tq[k] - m);
            float r = x - s_c[k * ENC_D + d];
            se += a;
            ser = fmaf(a, r, ser);
        }
        float e = ser / se;                 // E[b,n,d]
        Eb[d * ENC_N + n] = e;
        s_red[d * 65 + pix] = e;
    }
    __syncthreads();

    // deterministic fixed-order reduction over the 64 pixels of this tile
    if (t < ENC_D) {
        float s = 0.f;
#pragma unroll 8
        for (int p = 0; p < TILE_N; ++p) s += s_red[t * 65 + p];
        partial[(size_t)blockIdx.x * ENC_D + t] = s;
    }
}

// ---------------------------------------------------------------------------
// Pass 2: one wave32. Reduce tile partials -> EM/K, then gamma = sigmoid(
// EM @ fc_w^T + fc_b) via v_wmma_f32_16x16x32_f16 (M=16 pad of 8, N=64 in
// 4 tiles, K=64 in 2 steps -> 8 WMMA ops).
// ---------------------------------------------------------------------------
__global__ __launch_bounds__(32) void enc_pass2(
    const float* __restrict__ partial, const float* __restrict__ fc_w,
    const float* __restrict__ fc_b, float* __restrict__ gamma)
{
    __shared__ float s_em[ENC_B * ENC_D];   // EM mean, (8 x 64)

    const int lane = threadIdx.x;           // 0..31, all active -> EXEC all-1s
#pragma unroll
    for (int i = 0; i < 16; ++i) {
        const int idx = lane + i * 32;      // 512 entries / 32 lanes
        const int b = idx >> 6, d = idx & 63;
        float s = 0.f;
        for (int tile = 0; tile < TILES_PER_B; ++tile)
            s += partial[(size_t)(b * TILES_PER_B + tile) * ENC_D + d];
        s_em[idx] = s * (1.0f / (float)ENC_K);   // sum over N, mean over K
    }
    __syncthreads();

    const int m  = lane & 15;   // A: row M / B: col N within tile
    const int hi = lane >> 4;   // lane half selects K sub-block per ISA layout

    // A fragments: 16x32 f16, lanes 0-15: halves0-7=K0-7, halves8-15=K16-23;
    // lanes 16-31: halves0-7=K8-15, halves8-15=K24-31. Rows >= 8 are zero pad.
    v16h afrag[2];
#pragma unroll
    for (int kt = 0; kt < 2; ++kt) {
#pragma unroll
        for (int h = 0; h < 16; ++h) {
            const int kk = kt * 32 + ((h < 8) ? (hi * 8 + h) : (16 + hi * 8 + (h - 8)));
            const float v = (m < ENC_B) ? s_em[m * ENC_D + kk] : 0.f;
            afrag[kt][h] = (_Float16)v;
        }
    }

#pragma unroll
    for (int jt = 0; jt < 4; ++jt) {
        // B fragments: 32x16 f16, B[k][n] = fc_w[n_global*64 + k];
        // lanes 0-15: halves0-7=K0-7, halves8-15=K16-23; lanes 16-31: K8-15, K24-31
        v16h bfrag[2];
#pragma unroll
        for (int kt = 0; kt < 2; ++kt) {
#pragma unroll
            for (int h = 0; h < 16; ++h) {
                const int kk = kt * 32 + ((h < 8) ? (hi * 8 + h) : (16 + hi * 8 + (h - 8)));
                bfrag[kt][h] = (_Float16)fc_w[(jt * 16 + m) * ENC_D + kk];
            }
        }

        v8f c = {};
        c = __builtin_amdgcn_wmma_f32_16x16x32_f16(false, afrag[0], false, bfrag[0],
                                                   (short)0, c, false, false);
        c = __builtin_amdgcn_wmma_f32_16x16x32_f16(false, afrag[1], false, bfrag[1],
                                                   (short)0, c, false, false);

        // D layout: lanes 0-15 hold rows M=0..7 in c[0..7] (rows 8-15 are padding)
        if (hi == 0) {
            const int j = jt * 16 + m;
            const float bias = fc_b[j];
#pragma unroll
            for (int r = 0; r < ENC_B; ++r) {
                const float z = c[r] + bias;
                gamma[r * ENC_D + j] = 1.0f / (1.0f + __expf(-z));
            }
        }
    }
}

// ---------------------------------------------------------------------------
// Pass 3: out = relu(E * (1 + gamma[b,d])), in place on d_out, float4.
// ---------------------------------------------------------------------------
__global__ __launch_bounds__(256) void enc_pass3(
    float* __restrict__ out, const float* __restrict__ gamma)
{
    const int i   = blockIdx.x * 256 + threadIdx.x;   // float4 index
    const int row = i / (ENC_N / 4);                  // b*64 + d
    const float g = 1.0f + gamma[row];
    float4* o4 = reinterpret_cast<float4*>(out);
    float4 v = o4[i];
    v.x = fmaxf(v.x * g, 0.f);
    v.y = fmaxf(v.y * g, 0.f);
    v.z = fmaxf(v.z * g, 0.f);
    v.w = fmaxf(v.w * g, 0.f);
    o4[i] = v;
}

// ---------------------------------------------------------------------------
extern "C" void kernel_launch(void* const* d_in, const int* in_sizes, int n_in,
                              void* d_out, int out_size, void* d_ws, size_t ws_size,
                              hipStream_t stream) {
    (void)in_sizes; (void)n_in; (void)out_size; (void)ws_size;
    const float* X    = (const float*)d_in[0];
    const float* cw   = (const float*)d_in[1];
    const float* sc   = (const float*)d_in[2];
    const float* fc_w = (const float*)d_in[3];
    const float* fc_b = (const float*)d_in[4];
    float* out = (float*)d_out;

    float* partial = (float*)d_ws;                               // 392*64 floats
    float* gamma   = partial + ENC_B * TILES_PER_B * ENC_D;      // 512 floats

    enc_pass1<<<ENC_B * TILES_PER_B, 256, 0, stream>>>(X, cw, sc, out, partial);
    enc_pass2<<<1, 32, 0, stream>>>(partial, fc_w, fc_b, gamma);

    const int n4 = ENC_B * ENC_D * ENC_N / 4;                    // 401408
    enc_pass3<<<n4 / 256, 256, 0, stream>>>(out, gamma);
}
